// GlobalPointer_27633819582896
// MI455X (gfx1250) — matrix operand order repaired and checked by reading, not compiled, not run on hardware
//
#include <hip/hip_runtime.h>

#define BATCH 8
#define SEQ   512
#define HID   768
#define HEADS 52
#define DD    64
#define NOUT  (HEADS * 2 * DD)   /* 6656 */
#define MROWS (BATCH * SEQ)      /* 4096 */

typedef __attribute__((ext_vector_type(16))) __bf16 v16bf;
typedef __attribute__((ext_vector_type(8)))  __bf16 v8bf;
typedef __attribute__((ext_vector_type(8)))  float  v8f;
typedef int v4i __attribute__((vector_size(16)));   // pointee type of async-LDS builtins

#define HAS_ASYNC_LDS (__has_builtin(__builtin_amdgcn_global_load_async_to_lds_b128) && \
                       __has_builtin(__builtin_amdgcn_s_wait_asynccnt))

// ---- fragment loaders --------------------------------------------------
// A-matrix 16x32 bf16: lane L: m=L&15, hi=L>>4; K chunks [8hi,8hi+8) and [16+8hi,...)
__device__ __forceinline__ v16bf load_a_frag(const __bf16* __restrict__ base,
                                             int ld, int row0, int kt, int lane) {
  const int m  = row0 + (lane & 15);
  const int hi = lane >> 4;
  const __bf16* p = base + (size_t)m * ld + kt + hi * 8;
  v8bf lo = *(const v8bf*)(p);
  v8bf hh = *(const v8bf*)(p + 16);
  return __builtin_shufflevector(lo, hh, 0,1,2,3,4,5,6,7,8,9,10,11,12,13,14,15);
}

// Same A-fragment pattern but reading a 64-row LDS tile (ld = DD).
__device__ __forceinline__ v16bf load_a_frag_lds(const __bf16* sq, int row0, int kt, int lane) {
  const int m  = row0 + (lane & 15);
  const int hi = lane >> 4;
  const __bf16* p = sq + m * DD + kt + hi * 8;
  v8bf lo = *(const v8bf*)(p);
  v8bf hh = *(const v8bf*)(p + 16);
  return __builtin_shufflevector(lo, hh, 0,1,2,3,4,5,6,7,8,9,10,11,12,13,14,15);
}

// B-matrix 32x16 bf16: lane L holds column n=L&15, K chunks [8hi,8hi+8) and [16+8hi,...)
// For D = A x B with B = (Kmat)^T this is just row n of row-major Kmat.
__device__ __forceinline__ v16bf load_b_frag(const __bf16* __restrict__ base,
                                             int ld, int col0, int kt, int lane) {
  const int n  = col0 + (lane & 15);
  const int hi = lane >> 4;
  const __bf16* p = base + (size_t)n * ld + kt + hi * 8;
  v8bf lo = *(const v8bf*)(p);
  v8bf hh = *(const v8bf*)(p + 16);
  return __builtin_shufflevector(lo, hh, 0,1,2,3,4,5,6,7,8,9,10,11,12,13,14,15);
}

// ---- precision staging -------------------------------------------------
__global__ void __launch_bounds__(256) cvt_x_bf16(const float* __restrict__ X,
                                                  __bf16* __restrict__ Xb, int n) {
  int i = blockIdx.x * 256 + threadIdx.x;
  if (i < n) Xb[i] = (__bf16)X[i];
}

// W (HID x NOUT fp32, row-major) -> Wt (NOUT x HID bf16, row-major)
__global__ void __launch_bounds__(256) cvt_w_transpose(const float* __restrict__ W,
                                                       __bf16* __restrict__ Wt) {
  int id = blockIdx.x * 256 + threadIdx.x;   // id = n*HID + k  (write-coalesced)
  int n = id / HID;
  int k = id - n * HID;
  Wt[id] = (__bf16)W[(size_t)k * NOUT + n];
}

// ---- GEMM1: Y = X@W + b, fused RoPE, split into Q/K staging (bf16) ----
__global__ void __launch_bounds__(128) gemm_dense_rope(
    const __bf16* __restrict__ Xb, const __bf16* __restrict__ Wt,
    const float* __restrict__ bias, __bf16* __restrict__ Qw, __bf16* __restrict__ Kw) {
  const int lane = threadIdx.x & 31;
  const int wid  = blockIdx.x * 4 + (threadIdx.x >> 5);
  const int NT   = NOUT / 64;                 // 104 n-tiles
  const int m0   = (wid / NT) * 64;
  const int n0   = (wid % NT) * 64;

  v8f C[4][4] = {};
  for (int kt = 0; kt < HID; kt += 32) {
    if (kt + 32 < HID) {
      __builtin_prefetch(Xb + (size_t)(m0 + (lane & 15)) * HID + kt + 32, 0, 1);
      __builtin_prefetch(Wt + (size_t)(n0 + (lane & 15)) * HID + kt + 32, 0, 1);
    }
    v16bf a0 = load_a_frag(Xb, HID, m0,      kt, lane);
    v16bf a1 = load_a_frag(Xb, HID, m0 + 16, kt, lane);
    v16bf a2 = load_a_frag(Xb, HID, m0 + 32, kt, lane);
    v16bf a3 = load_a_frag(Xb, HID, m0 + 48, kt, lane);
#pragma unroll
    for (int jn = 0; jn < 4; ++jn) {
      v16bf bf = load_b_frag(Wt, HID, n0 + 16 * jn, kt, lane);
      C[0][jn] = __builtin_amdgcn_wmma_f32_16x16x32_bf16(false, a0, false, bf, (short)0, C[0][jn], false, false);
      C[1][jn] = __builtin_amdgcn_wmma_f32_16x16x32_bf16(false, a1, false, bf, (short)0, C[1][jn], false, false);
      C[2][jn] = __builtin_amdgcn_wmma_f32_16x16x32_bf16(false, a2, false, bf, (short)0, C[2][jn], false, false);
      C[3][jn] = __builtin_amdgcn_wmma_f32_16x16x32_bf16(false, a3, false, bf, (short)0, C[3][jn], false, false);
    }
  }

  // Epilogue: +bias, RoPE (interleaved), store bf16 into [b][h][s][d] staging.
  const int nl  = lane & 15;
  const int hiq = lane >> 4;
#pragma unroll
  for (int jn = 0; jn < 4; ++jn) {
    const int n  = n0 + 16 * jn + nl;
    const int h  = n >> 7;                    // head = n / 128
    const int c  = n & 127;                   // within head: [0,64)=q, [64,128)=k
    const int d  = c & 63;
    __bf16* dst  = (c < 64) ? Qw : Kw;
    const float bi  = bias[n];
    // inv_freq = 10000^(-(2*(d/2))/64) = exp(-(ln 1e4)/64 * (d & ~1))
    const float inv = __expf(-(9.210340371976184f / 64.0f) * (float)(d & ~1));
    const float sgn = (d & 1) ? 1.0f : -1.0f; // rot[2i]=-x[2i+1], rot[2i+1]=x[2i]
#pragma unroll
    for (int im = 0; im < 4; ++im) {
#pragma unroll
      for (int r = 0; r < 8; ++r) {
        const int m  = m0 + 16 * im + r + 8 * hiq;
        const int bb = m >> 9;                // batch
        const int s  = m & (SEQ - 1);         // position
        float y = C[im][jn][r] + bi;
        float partner = __shfl_xor(y, 1, 32); // value at d^1 (bias included)
        float ang = (float)s * inv;
        float v = y * cosf(ang) + sgn * partner * sinf(ang);
        dst[(((size_t)(bb * HEADS + h)) * SEQ + s) * DD + d] = (__bf16)v;
      }
    }
  }
}

// ---- Kernel 2: per-(b,h) logits = q @ k^T, mask + tril + scale --------
// All 4 waves of a block share one (bh, m-block): the 64x64 Q tile is staged
// once into LDS with async global->LDS copies (ASYNCcnt path), then each wave
// computes a distinct 64-wide n-block against it.
__global__ void __launch_bounds__(128) logits_wmma(
    const __bf16* __restrict__ Qw, const __bf16* __restrict__ Kw,
    const int* __restrict__ amask, float* __restrict__ out) {
  __shared__ __bf16 sq[64 * DD];              // 8 KB: Q rows m0..m0+63, ld=64
  const int tid  = threadIdx.x;
  const int lane = tid & 31;
  const int wave = tid >> 5;
  const int bh   = blockIdx.x >> 4;           // 416 (b,h) pairs
  const int sub  = blockIdx.x & 15;
  const int m0   = (sub >> 1) * 64;
  const int n0   = ((sub & 1) * 4 + wave) * 64;
  const int b    = bh / HEADS;
  const __bf16* Qb = Qw + (size_t)bh * SEQ * DD;
  const __bf16* Kb = Kw + (size_t)bh * SEQ * DD;

  // ---- cooperative Q-tile staging: 512 chunks x 16B = 8 KB ----
#if HAS_ASYNC_LDS
#pragma unroll
  for (int i = 0; i < 4; ++i) {
    const int chunk = tid + 128 * i;
    const int row   = chunk >> 3;
    const int col   = (chunk & 7) * 8;        // bf16 elements
    __builtin_amdgcn_global_load_async_to_lds_b128(
        (v4i*)(Qb + (size_t)(m0 + row) * DD + col),
        (v4i*)&sq[row * DD + col],
        0, 0);
  }
  __builtin_amdgcn_s_wait_asynccnt(0);
  __syncthreads();
#else
#pragma unroll
  for (int i = 0; i < 4; ++i) {
    const int chunk = tid + 128 * i;
    const int row   = chunk >> 3;
    const int col   = (chunk & 7) * 8;
    *(v8bf*)&sq[row * DD + col] = *(const v8bf*)(Qb + (size_t)(m0 + row) * DD + col);
  }
  __syncthreads();
#endif

  v8f C[4][4] = {};
#pragma unroll
  for (int kt = 0; kt < DD; kt += 32) {
    v16bf a0 = load_a_frag_lds(sq,  0, kt, lane);
    v16bf a1 = load_a_frag_lds(sq, 16, kt, lane);
    v16bf a2 = load_a_frag_lds(sq, 32, kt, lane);
    v16bf a3 = load_a_frag_lds(sq, 48, kt, lane);
#pragma unroll
    for (int jn = 0; jn < 4; ++jn) {
      v16bf bf = load_b_frag(Kb, DD, n0 + 16 * jn, kt, lane);
      C[0][jn] = __builtin_amdgcn_wmma_f32_16x16x32_bf16(false, a0, false, bf, (short)0, C[0][jn], false, false);
      C[1][jn] = __builtin_amdgcn_wmma_f32_16x16x32_bf16(false, a1, false, bf, (short)0, C[1][jn], false, false);
      C[2][jn] = __builtin_amdgcn_wmma_f32_16x16x32_bf16(false, a2, false, bf, (short)0, C[2][jn], false, false);
      C[3][jn] = __builtin_amdgcn_wmma_f32_16x16x32_bf16(false, a3, false, bf, (short)0, C[3][jn], false, false);
    }
  }

  const int nl  = lane & 15;
  const int hiq = lane >> 4;
  float* ob = out + (size_t)bh * SEQ * SEQ;
#pragma unroll
  for (int jn = 0; jn < 4; ++jn) {
    const int n = n0 + 16 * jn + nl;
    const float pad = (float)amask[b * SEQ + n];
    const float neg = -(1.0f - pad) * 1.0e12f;
#pragma unroll
    for (int im = 0; im < 4; ++im) {
#pragma unroll
      for (int r = 0; r < 8; ++r) {
        const int m = m0 + 16 * im + r + 8 * hiq;
        float v = C[im][jn][r] * pad + neg;
        if (m > n) v -= 1.0e12f;              // tril(ones,-1) causal-style mask
        ob[(size_t)m * SEQ + n] = v * 0.125f; // / sqrt(64)
      }
    }
  }
}

// ---- launch ------------------------------------------------------------
extern "C" void kernel_launch(void* const* d_in, const int* in_sizes, int n_in,
                              void* d_out, int out_size, void* d_ws, size_t ws_size,
                              hipStream_t stream) {
  const float* X     = (const float*)d_in[0];   // (8,512,768) fp32
  const int*   amask = (const int*)d_in[1];     // (8,512) int32
  const float* Wd    = (const float*)d_in[2];   // (768,6656) fp32
  const float* bd    = (const float*)d_in[3];   // (6656,) fp32
  float* out = (float*)d_out;                   // (8,52,512,512) fp32

  char* ws = (char*)d_ws;
  __bf16* Xb = (__bf16*)(ws);                          // 4096*768*2   = 6,291,456
  __bf16* Wt = (__bf16*)(ws + 6291456);                // 6656*768*2   = 10,223,616
  __bf16* Qw = (__bf16*)(ws + 16515072);               // 8*52*512*64*2 = 27,262,976
  __bf16* Kw = (__bf16*)(ws + 43778048);               // same          -> 71,041,024 total

  const int nX = MROWS * HID;                   // 3,145,728
  const int nW = NOUT * HID;                    // 5,111,808
  cvt_x_bf16<<<nX / 256, 256, 0, stream>>>(X, Xb, nX);
  cvt_w_transpose<<<nW / 256, 256, 0, stream>>>(Wd, Wt);

  // GEMM1: 64x64 tile per wave, 64*104 tiles, 4 waves/block
  gemm_dense_rope<<<(64 * (NOUT / 64)) / 4, 128, 0, stream>>>(Xb, Wt, bd, Qw, Kw);

  // Logits: 416 (b,h) blocks * 16 sub-blocks; 4 waves share one Q tile in LDS
  logits_wmma<<<416 * 16, 128, 0, stream>>>(Qw, Kw, amask, out);
}
